// _ConvertedBidirectionalBandRNN_19490561589978
// MI455X (gfx1250) — matrix operand order repaired and checked by reading, not compiled, not run on hardware
//
#include <hip/hip_runtime.h>

typedef _Float16 half_t;
typedef __attribute__((ext_vector_type(16))) _Float16 v16h;
typedef __attribute__((ext_vector_type(8)))  _Float16 v8h;
typedef __attribute__((ext_vector_type(8)))  float    v8f;

#define B_   16
#define T_   512
#define NB_  32
#define BI_  64
#define BH_  64
#define ROWS (B_ * T_)        // 8192
#define XC   (NB_ * BI_)      // 2048 input features per row
#define OC   (NB_ * 2 * BH_)  // 4096 output features per row
#define G_   (3 * BH_)        // 192 gate rows

__device__ __forceinline__ float sigmoid_f(float x) {
    return 1.0f / (1.0f + __expf(-x));
}
__device__ __forceinline__ float tanh_f(float x) {
    return 2.0f / (1.0f + __expf(-2.0f * x)) - 1.0f;
}

__device__ __forceinline__ v8f wmma16(v16h a, v16h b, v8f c) {
    // v_wmma_f32_16x16x32_f16: D = A(16x32) * B(32x16) + C(16x16 f32)
    return __builtin_amdgcn_wmma_f32_16x16x32_f16(
        /*neg_a=*/false, a, /*neg_b=*/false, b,
        /*c_mod=*/(short)0, c, /*reuse_a=*/false, /*reuse_b=*/false);
}

// A-fragment (16x32, f16) from row-major fp32 source; row stride ld, K base kc.
// ISA layout: lane L<16 -> row L, K in {kc+0..7, kc+16..23}; lane 16+L -> K {kc+8..15, kc+24..31}.
__device__ __forceinline__ v16h load_a_f32(const float* __restrict__ src, int ld,
                                           int kc, int lane) {
    const int m  = lane & 15;
    const int k0 = (lane >> 4) << 3;   // 0 or 8
    const float* p = src + (size_t)m * ld + kc + k0;
    const float4 f0 = *reinterpret_cast<const float4*>(p);
    const float4 f1 = *reinterpret_cast<const float4*>(p + 4);
    const float4 f2 = *reinterpret_cast<const float4*>(p + 16);
    const float4 f3 = *reinterpret_cast<const float4*>(p + 20);
    v16h a;
    a[0]  = (half_t)f0.x; a[1]  = (half_t)f0.y; a[2]  = (half_t)f0.z; a[3]  = (half_t)f0.w;
    a[4]  = (half_t)f1.x; a[5]  = (half_t)f1.y; a[6]  = (half_t)f1.z; a[7]  = (half_t)f1.w;
    a[8]  = (half_t)f2.x; a[9]  = (half_t)f2.y; a[10] = (half_t)f2.z; a[11] = (half_t)f2.w;
    a[12] = (half_t)f3.x; a[13] = (half_t)f3.y; a[14] = (half_t)f3.z; a[15] = (half_t)f3.w;
    return a;
}

// A-fragment from f16 LDS (row stride ld elements).
__device__ __forceinline__ v16h load_a_f16(const half_t* __restrict__ src, int ld,
                                           int kc, int lane) {
    const int m  = lane & 15;
    const int k0 = (lane >> 4) << 3;
    const half_t* p = src + m * ld + kc + k0;
    const v8h lo = *reinterpret_cast<const v8h*>(p);       // 16B ds_load
    const v8h hi = *reinterpret_cast<const v8h*>(p + 16);  // 16B ds_load
    v16h a;
#pragma unroll
    for (int j = 0; j < 8; ++j) { a[j] = lo[j]; a[8 + j] = hi[j]; }
    return a;
}

// B-fragment (32x16, f16) for K-chunk kc from w[out][in] row-major fp32 (in-dim = BH_).
// ISA layout: lane holds column N = lane&15; K = kc + (lane>>4)*8 + {0..7, 16..23}.
__device__ __forceinline__ v16h load_b_w(const float* __restrict__ w, int n_off,
                                         int kc, int lane) {
    const int n  = lane & 15;
    const int kb = kc + ((lane >> 4) << 3);
    const float* p = w + (size_t)(n_off + n) * BH_ + kb;
    const float4 f0 = *reinterpret_cast<const float4*>(p);
    const float4 f1 = *reinterpret_cast<const float4*>(p + 4);
    const float4 f2 = *reinterpret_cast<const float4*>(p + 16);
    const float4 f3 = *reinterpret_cast<const float4*>(p + 20);
    v16h b;
    b[0]  = (half_t)f0.x; b[1]  = (half_t)f0.y; b[2]  = (half_t)f0.z; b[3]  = (half_t)f0.w;
    b[4]  = (half_t)f1.x; b[5]  = (half_t)f1.y; b[6]  = (half_t)f1.z; b[7]  = (half_t)f1.w;
    b[8]  = (half_t)f2.x; b[9]  = (half_t)f2.y; b[10] = (half_t)f2.z; b[11] = (half_t)f2.w;
    b[12] = (half_t)f3.x; b[13] = (half_t)f3.y; b[14] = (half_t)f3.z; b[15] = (half_t)f3.w;
    return b;
}

__global__ __launch_bounds__(128)
void band_gru_bidir_kernel(const float* __restrict__ x,
                           const float* __restrict__ w_ih_f, const float* __restrict__ w_hh_f,
                           const float* __restrict__ b_ih_f, const float* __restrict__ b_hh_f,
                           const float* __restrict__ w_ih_r, const float* __restrict__ w_hh_r,
                           const float* __restrict__ b_ih_r, const float* __restrict__ b_hh_r,
                           float* __restrict__ out) {
    __shared__ __align__(16) half_t h_lds[16 * BH_];   // 2 KB: 16 rows x 64 hidden, f16

    const int tid     = threadIdx.x;
    const int wave    = tid >> 5;
    const int lane    = tid & 31;
    const int row0    = blockIdx.x * 16;
    const int n_base  = wave * 16;           // this wave's hidden-column slice
    const int half_id = lane >> 4;
    const int col     = n_base + (lane & 15);

#pragma unroll 1
    for (int dir = 0; dir < 2; ++dir) {
        const float* w_ih = dir ? w_ih_r : w_ih_f;
        const float* w_hh = dir ? w_hh_r : w_hh_f;
        const float* b_ih = dir ? b_ih_r : b_ih_f;
        const float* b_hh = dir ? b_hh_r : b_hh_f;

        // Per-lane gate biases for this hidden column.
        float bi[3], bh[3];
#pragma unroll
        for (int g = 0; g < 3; ++g) {
            bi[g] = b_ih[g * BH_ + col];
            bh[g] = b_hh[g * BH_ + col];
        }

        // Preload weight B-fragments for this wave's 16 columns of each gate.
        v16h Bi[3][2], Bh[3][2];
#pragma unroll
        for (int g = 0; g < 3; ++g) {
#pragma unroll
            for (int c = 0; c < 2; ++c) {
                Bi[g][c] = load_b_w(w_ih, g * BH_ + n_base, c * 32, lane);
                Bh[g][c] = load_b_w(w_hh, g * BH_ + n_base, c * 32, lane);
            }
        }

        // h = 0 (registers, matching C-fragment layout) and zero the LDS copy.
        v8f hprev = {};
        for (int i = tid; i < 16 * BH_; i += 128) h_lds[i] = (half_t)0.0f;
        __syncthreads();

#pragma unroll 1
        for (int t = 0; t < NB_; ++t) {
            const int bt = dir ? (NB_ - 1 - t) : t;

            // A fragments: x band (global fp32 -> f16) and hidden state (LDS f16).
            const float* xblk = x + (size_t)row0 * XC + (size_t)bt * BI_;
            const v16h Ax0 = load_a_f32(xblk, XC, 0,  lane);
            const v16h Ax1 = load_a_f32(xblk, XC, 32, lane);
            const v16h Ah0 = load_a_f16(h_lds, BH_, 0,  lane);
            const v16h Ah1 = load_a_f16(h_lds, BH_, 32, lane);

            // r and z gates: gi and gh accumulate into the same f32 accumulator.
            v8f zero = {};
            v8f acc_r = wmma16(Ax0, Bi[0][0], zero);
            acc_r     = wmma16(Ax1, Bi[0][1], acc_r);
            acc_r     = wmma16(Ah0, Bh[0][0], acc_r);
            acc_r     = wmma16(Ah1, Bh[0][1], acc_r);

            v8f acc_z = wmma16(Ax0, Bi[1][0], zero);
            acc_z     = wmma16(Ax1, Bi[1][1], acc_z);
            acc_z     = wmma16(Ah0, Bh[1][0], acc_z);
            acc_z     = wmma16(Ah1, Bh[1][1], acc_z);

            // n gate: keep input and hidden parts separate (r scales only gh_n + b_hh_n).
            v8f gin = wmma16(Ax0, Bi[2][0], zero);
            gin     = wmma16(Ax1, Bi[2][1], gin);
            v8f ghn = wmma16(Ah0, Bh[2][0], zero);
            ghn     = wmma16(Ah1, Bh[2][1], ghn);

            __syncthreads();   // every wave finished reading h_lds for this step

            // Elementwise GRU update in fp32; write new h to LDS (f16) and output (f32).
#pragma unroll
            for (int v = 0; v < 8; ++v) {
                const float r  = sigmoid_f(acc_r[v] + bi[0] + bh[0]);
                const float zg = sigmoid_f(acc_z[v] + bi[1] + bh[1]);
                const float nn = tanh_f(gin[v] + bi[2] + r * (ghn[v] + bh[2]));
                const float hn = (1.0f - zg) * nn + zg * hprev[v];
                hprev[v] = hn;

                const int mrow = v + half_id * 8;      // C-fragment row for this VGPR
                h_lds[mrow * BH_ + col] = (half_t)hn;
                out[(size_t)(row0 + mrow) * OC + (size_t)bt * (2 * BH_) + dir * BH_ + col] = hn;
            }
            __syncthreads();   // new h visible before next step's fragment loads
        }
        __syncthreads();       // safe re-zero of h_lds for the reverse pass
    }
}

extern "C" void kernel_launch(void* const* d_in, const int* in_sizes, int n_in,
                              void* d_out, int out_size, void* d_ws, size_t ws_size,
                              hipStream_t stream) {
    const float* x      = (const float*)d_in[0];
    const float* w_ih_f = (const float*)d_in[1];
    const float* w_hh_f = (const float*)d_in[2];
    const float* b_ih_f = (const float*)d_in[3];
    const float* b_hh_f = (const float*)d_in[4];
    const float* w_ih_r = (const float*)d_in[5];
    const float* w_hh_r = (const float*)d_in[6];
    const float* b_ih_r = (const float*)d_in[7];
    const float* b_hh_r = (const float*)d_in[8];
    float* out = (float*)d_out;

    dim3 grid(ROWS / 16);   // 512 workgroups, one 16-row tile each
    dim3 block(128);        // 4 waves: each owns 16 hidden columns
    band_gru_bidir_kernel<<<grid, block, 0, stream>>>(
        x, w_ih_f, w_hh_f, b_ih_f, b_hh_f,
        w_ih_r, w_hh_r, b_ih_r, b_hh_r, out);
}
